// EdgeMPNNLayer_84301618086097
// MI455X (gfx1250) — compile-verified, matrix operand order checked by reading, not run on hardware
//
#include <hip/hip_runtime.h>
#include <math.h>

// ---------------------------------------------------------------------------
// EdgeMPNN layer for MI455X (gfx1250, wave32, WMMA).
//  - GEMMs on v_wmma_f32_16x16x32_f16 (f16 in, fp32 accumulate)
//  - h pre-converted to f16 (gathered ~20x per row -> convert once)
//  - weights pre-swizzled to B-fragment layout in ws; edge kernel stages the
//    contiguous e1|e2|src block (172032 B) into LDS via CDNA5 async-to-LDS DMA
//    (global_load_async_to_lds_b128 + s_wait_asynccnt), amortized by a
//    grid-stride loop over edge tiles.
//  - agg built with global f32 scatter-atomics (L2-resident; h fits in L2)
// ---------------------------------------------------------------------------

typedef __attribute__((ext_vector_type(16))) _Float16 v16h;
typedef __attribute__((ext_vector_type(8)))  _Float16 v8h;
typedef __attribute__((ext_vector_type(8)))  float    v8f;

#define H   128
#define ED  16
#define NN  50000
#define EE  500000
#define EPSV 1e-5f

#define EWAVES 8   // waves per block, edge kernel (256 threads)
#define NWAVES 4   // waves per block, node kernel

// D = A*B + C, 16x16x32 f16 -> f32
#define WMMA(a, b, c) \
  __builtin_amdgcn_wmma_f32_16x16x32_f16(false, (a), false, (b), (short)0, (c), false, false)

__device__ __forceinline__ float gelu_exact(float x) {
  return 0.5f * x * (1.f + erff(x * 0.70710678118654752f));
}
__device__ __forceinline__ float sigmoidf_(float x) {
  return 1.f / (1.f + expf(-x));
}

// ---- fragment helpers -----------------------------------------------------
// A/B 16-bit fragment k-mapping (ISA 7.12.2): for lane l, half i:
//   k = (i&7) + (i>=8 ? 16 : 0) + (l>=16 ? 8 : 0)   (within a 32-wide K tile)
//   A: lane%16 = M row,  B: lane%16 = N col.

__device__ __forceinline__ v16h b_frag(const _Float16* __restrict__ w, int tile, int lane) {
  return *(const v16h*)(w + (((size_t)tile * 32 + lane) << 4));
}

// A-fragment from an f16 row (global h16 copy or LDS hid tile): 2x 16B loads
__device__ __forceinline__ v16h a_frag_h(const _Float16* __restrict__ row, int ktbase, int lane) {
  const int c0 = ktbase + ((lane & 16) ? 8 : 0);
  const v8h lo = *(const v8h*)(row + c0);
  const v8h hi = *(const v8h*)(row + c0 + 16);
  v16h a;
#pragma unroll
  for (int i = 0; i < 8; ++i) { a[i] = lo[i]; a[8 + i] = hi[i]; }
  return a;
}

__device__ __forceinline__ v16h a_frag_f32_scaled(const float* __restrict__ row, int ktbase,
                                                  int lane, float s) {
  const int c0 = ktbase + ((lane & 16) ? 8 : 0);
  v16h a;
#pragma unroll
  for (int i = 0; i < 8; ++i) a[i] = (_Float16)(row[c0 + i] * s);
#pragma unroll
  for (int i = 0; i < 8; ++i) a[8 + i] = (_Float16)(row[c0 + 16 + i] * s);
  return a;
}

// ---- weight swizzle: fp32 row-major (K x Nn) -> f16 B-fragment tiles -------
__global__ void k_swizzle(const float* __restrict__ W, _Float16* __restrict__ out,
                          int K, int Kpad, int Nn) {
  const int id = blockIdx.x * blockDim.x + threadIdx.x;
  const int NT = Nn >> 4;
  const int total = (Kpad >> 5) * NT * 512;
  if (id >= total) return;
  const int i = id & 15;
  const int l = (id >> 4) & 31;
  const int t = id >> 9;
  const int kt = t / NT, nt = t - kt * NT;
  const int k = kt * 32 + (i & 7) + ((i & 8) ? 16 : 0) + ((l & 16) ? 8 : 0);
  const int n = nt * 16 + (l & 15);
  const float v = (k < K) ? W[(size_t)k * Nn + n] : 0.f;
  out[id] = (_Float16)v;
}

__global__ void k_h16(const float* __restrict__ h, _Float16* __restrict__ h16, size_t total) {
  size_t id = (size_t)blockIdx.x * blockDim.x + threadIdx.x;
  if (id < total) h16[id] = (_Float16)h[id];
}

__global__ void k_zero(float* __restrict__ p, size_t total) {
  size_t id = (size_t)blockIdx.x * blockDim.x + threadIdx.x;
  if (id < total) p[id] = 0.f;
}

__global__ void k_indeg(const int* __restrict__ eidx, float* __restrict__ indeg, int E) {
  int e = blockIdx.x * blockDim.x + threadIdx.x;
  if (e < E) atomicAdd(&indeg[eidx[2 * e + 1]], 1.f);
}

// ---- edge kernel: grid-stride, 16 edges per wave per iteration -------------
__global__ __launch_bounds__(32 * EWAVES) void k_edges(
    const float* __restrict__ ea, const int* __restrict__ eidx,
    const _Float16* __restrict__ h16,
    const _Float16* __restrict__ wglob,   // contiguous e1|e2|src fragments, 172032 B
    const float* __restrict__ e1b, const float* __restrict__ e2b,
    const float* __restrict__ srcb,
    float* __restrict__ agg, int ntiles) {
  __shared__ alignas(32) _Float16 s_w[86016];          // 172032 B of weights
  __shared__ alignas(32) _Float16 s_hid[EWAVES][16 * H];
  __shared__ int s_dst[EWAVES][16];

  // ---- one-time staging of weight fragments: CDNA5 async DMA to LDS ----
  {
    const unsigned lbase = (unsigned)(uintptr_t)(&s_w[0]);
    const unsigned long long gbase = (unsigned long long)(uintptr_t)wglob;
    for (int i = threadIdx.x; i < 172032 / 16; i += 32 * EWAVES) {
      const unsigned laddr = lbase + (unsigned)i * 16u;
      const unsigned long long ga = gbase + (unsigned long long)i * 16ull;
      asm volatile("global_load_async_to_lds_b128 %0, %1, off"
                   :: "v"(laddr), "v"(ga) : "memory");
    }
    asm volatile("s_wait_asynccnt 0x0" ::: "memory");
  }
  __syncthreads();   // uniform: every thread executes this exactly once

  const _Float16* we1  = s_w;           // 288x128  (9 kt x 8 nt tiles)
  const _Float16* we2  = s_w + 36864;   // 128x256  (4 kt x 16 nt tiles)
  const _Float16* wsrc = s_w + 69632;   // 128x128  (4 kt x 8 nt tiles)

  const int lane = threadIdx.x & 31;
  const int wv   = threadIdx.x >> 5;
  const int m    = lane & 15;
  const int loff = (lane & 16) ? 8 : 0;
  const int stride = gridDim.x * EWAVES;

  for (int tile = blockIdx.x * EWAVES + wv; tile < ntiles; tile += stride) {
    const int e0 = tile << 4;
    const int2 se = ((const int2*)eidx)[e0 + m];      // (src, dst) of edge-row m
    if (lane < 16) s_dst[wv][m] = se.y;
    const _Float16* hs = h16 + (size_t)se.x * H;
    const _Float16* hd = h16 + (size_t)se.y * H;
    v16h a_s[4], a_d[4];
#pragma unroll
    for (int kt = 0; kt < 4; ++kt) {
      a_s[kt] = a_frag_h(hs, kt * 32, lane);
      a_d[kt] = a_frag_h(hd, kt * 32, lane);
    }
    // edge_attr fragment: K tile 8 of edge_input (cols 256..271, zero-pad to 287)
    v16h a_e;
#pragma unroll
    for (int i = 0; i < 16; ++i) a_e[i] = (_Float16)0.f;
    const float* erow = ea + (size_t)(e0 + m) * ED + ((lane & 16) ? 8 : 0);
#pragma unroll
    for (int i = 0; i < 8; ++i) a_e[i] = (_Float16)erow[i];

    // hid = gelu(edge_input @ e1_W + e1_b)  ->  LDS (f16, row-major tile)
#pragma unroll
    for (int nt = 0; nt < 8; ++nt) {
      v8f acc = {};
#pragma unroll
      for (int kt = 0; kt < 4; ++kt)
        acc = WMMA(a_s[kt], b_frag(we1, kt * 8 + nt, lane), acc);
#pragma unroll
      for (int kt = 0; kt < 4; ++kt)
        acc = WMMA(a_d[kt], b_frag(we1, (kt + 4) * 8 + nt, lane), acc);
      acc = WMMA(a_e, b_frag(we1, 8 * 8 + nt, lane), acc);
      const float b = e1b[nt * 16 + m];
#pragma unroll
      for (int r = 0; r < 8; ++r)
        s_hid[wv][(r + loff) * H + nt * 16 + m] = (_Float16)gelu_exact(acc[r] + b);
    }
    // intra-wave D->A re-fragmentation: same-wave LDS ops are in-order (DScnt);
    // only compiler reordering must be fenced.
    asm volatile("" ::: "memory");

    v16h a_h[4];
    const _Float16* hrow = &s_hid[wv][m * H];
#pragma unroll
    for (int kt = 0; kt < 4; ++kt) a_h[kt] = a_frag_h(hrow, kt * 32, lane);

#pragma unroll
    for (int nt = 0; nt < 8; ++nt) {
      v8f g = {}, sh = {}, sl = {};
#pragma unroll
      for (int kt = 0; kt < 4; ++kt) {
        g  = WMMA(a_h[kt], b_frag(we2, kt * 16 + nt,     lane), g);   // gate
        sh = WMMA(a_h[kt], b_frag(we2, kt * 16 + nt + 8, lane), sh);  // shift
        sl = WMMA(a_s[kt], b_frag(wsrc, kt * 8 + nt,     lane), sl);  // h_src@src_W
      }
      const int col = nt * 16 + m;
      const float gb = e2b[col], shb = e2b[H + col], slb = srcb[col];
#pragma unroll
      for (int r = 0; r < 8; ++r) {
        const float msg = sigmoidf_(g[r] + gb) * (sl[r] + slb) + (sh[r] + shb);
        const int dn = s_dst[wv][r + loff];
        atomicAdd(&agg[(size_t)dn * H + col], msg);
      }
    }
    asm volatile("" ::: "memory");   // WAR fence before next iteration's stores
  }
}

// ---- node kernel: 16 nodes per wave, fused update + residual + LayerNorm --
__global__ __launch_bounds__(32 * NWAVES) void k_nodes(
    const float* __restrict__ h, const _Float16* __restrict__ h16,
    const float* __restrict__ agg, const float* __restrict__ indeg,
    const _Float16* __restrict__ selfw, const _Float16* __restrict__ aggw,
    const float* __restrict__ selfb, const float* __restrict__ aggb,
    const float* __restrict__ lng, const float* __restrict__ lnb,
    float* __restrict__ out, int ntiles) {
  __shared__ alignas(32) float s_x[NWAVES][16 * H];
  const int lane = threadIdx.x & 31;
  const int wv   = threadIdx.x >> 5;
  const int tile = blockIdx.x * NWAVES + wv;
  const int m    = lane & 15;
  const int loff = (lane & 16) ? 8 : 0;
  const int n0   = tile << 4;

  if (tile < ntiles) {
    const int node = n0 + m;
    const _Float16* hrow = h16 + (size_t)node * H;
    const float* grow = agg + (size_t)node * H;
    const float inv = 1.f / fmaxf(indeg[node], 1.f);
    v16h a_hn[4], a_g[4];
#pragma unroll
    for (int kt = 0; kt < 4; ++kt) {
      a_hn[kt] = a_frag_h(hrow, kt * 32, lane);
      a_g[kt]  = a_frag_f32_scaled(grow, kt * 32, lane, inv);
    }
#pragma unroll
    for (int nt = 0; nt < 8; ++nt) {
      v8f acc = {};
#pragma unroll
      for (int kt = 0; kt < 4; ++kt) {
        acc = WMMA(a_hn[kt], b_frag(selfw, kt * 8 + nt, lane), acc);
        acc = WMMA(a_g[kt],  b_frag(aggw,  kt * 8 + nt, lane), acc);
      }
      const int col = nt * 16 + m;
      const float b = selfb[col] + aggb[col];
#pragma unroll
      for (int r = 0; r < 8; ++r) {
        const int mr = r + loff;
        const float hv = h[(size_t)(n0 + mr) * H + col];   // fp32 residual
        s_x[wv][mr * H + col] = hv + gelu_exact(acc[r] + b);
      }
    }
  }
  __syncthreads();
  if (tile < ntiles) {
    // LayerNorm: row m handled by lane pair (l, l+16), 64 cols each
    const int cb = (lane & 16) ? 64 : 0;
    const float* xr = &s_x[wv][m * H];
    float s = 0.f, ss = 0.f;
#pragma unroll
    for (int c = 0; c < 64; ++c) { const float v = xr[cb + c]; s += v; ss += v * v; }
    s  += __shfl_xor(s, 16);
    ss += __shfl_xor(ss, 16);
    const float mu   = s * (1.f / H);
    const float var  = ss * (1.f / H) - mu * mu;
    const float rstd = rsqrtf(var + EPSV);
    float* orow = out + (size_t)(n0 + m) * H;
#pragma unroll 8
    for (int c = 0; c < 64; ++c) {
      const int cc = cb + c;
      orow[cc] = (xr[cc] - mu) * rstd * lng[cc] + lnb[cc];
    }
  }
}

// ---------------------------------------------------------------------------
extern "C" void kernel_launch(void* const* d_in, const int* in_sizes, int n_in,
                              void* d_out, int out_size, void* d_ws, size_t ws_size,
                              hipStream_t stream) {
  (void)in_sizes; (void)n_in; (void)out_size; (void)ws_size;
  const float* h     = (const float*)d_in[0];
  const float* ea    = (const float*)d_in[1];
  const int*   eidx  = (const int*)d_in[2];
  const float* srcW  = (const float*)d_in[3];
  const float* srcb  = (const float*)d_in[4];
  const float* e1W   = (const float*)d_in[5];
  const float* e1b   = (const float*)d_in[6];
  const float* e2W   = (const float*)d_in[7];
  const float* e2b   = (const float*)d_in[8];
  const float* selfW = (const float*)d_in[9];
  const float* selfb = (const float*)d_in[10];
  const float* aggW  = (const float*)d_in[11];
  const float* aggb  = (const float*)d_in[12];
  const float* lng   = (const float*)d_in[13];
  const float* lnb   = (const float*)d_in[14];

  // workspace carve-out (256B-aligned); ~38.9 MB total
  char* ws = (char*)d_ws;
  _Float16* e1h   = (_Float16*)(ws + 0);        // 288x128 f16 (K padded 272->288)
  _Float16* e2h   = (_Float16*)(ws + 73728);    // 128x256 f16
  _Float16* srch  = (_Float16*)(ws + 139264);   // 128x128 f16   (e1|e2|src contiguous)
  _Float16* selfh = (_Float16*)(ws + 172032);   // 128x128 f16
  _Float16* aggh  = (_Float16*)(ws + 204800);   // 128x128 f16
  float*    agg   = (float*)(ws + 237568);                          // N x 128 f32
  float*    indeg = (float*)(ws + 237568 + (size_t)NN * H * 4);     // N f32
  _Float16* h16   = (_Float16*)(ws + 26037760);                     // N x 128 f16

  // 1) swizzle weights into B-fragment layout; convert h to f16
  k_swizzle<<<(9 * 8 * 512 + 255) / 256, 256, 0, stream>>>(e1W, e1h, 272, 288, 128);
  k_swizzle<<<(4 * 16 * 512 + 255) / 256, 256, 0, stream>>>(e2W, e2h, 128, 128, 256);
  k_swizzle<<<(4 * 8 * 512 + 255) / 256, 256, 0, stream>>>(srcW, srch, 128, 128, 128);
  k_swizzle<<<(4 * 8 * 512 + 255) / 256, 256, 0, stream>>>(selfW, selfh, 128, 128, 128);
  k_swizzle<<<(4 * 8 * 512 + 255) / 256, 256, 0, stream>>>(aggW, aggh, 128, 128, 128);
  const size_t htot = (size_t)NN * H;
  k_h16<<<(int)((htot + 255) / 256), 256, 0, stream>>>(h, h16, htot);

  // 2) zero agg + indeg (contiguous)
  const size_t ztot = (size_t)NN * H + NN;
  k_zero<<<(int)((ztot + 255) / 256), 256, 0, stream>>>(agg, ztot);

  // 3) in-degrees
  k_indeg<<<(EE + 255) / 256, 256, 0, stream>>>(eidx, indeg, EE);

  // 4) edge GEMMs + message scatter (E % 16 == 0), grid-stride over tiles
  const int etiles = EE / 16;
  int eblocks = (etiles + EWAVES - 1) / EWAVES;
  if (eblocks > 768) eblocks = 768;
  k_edges<<<eblocks, 32 * EWAVES, 0, stream>>>(
      ea, eidx, h16, e1h, e1b, e2b, srcb, agg, etiles);

  // 5) node GEMMs + gelu residual + LayerNorm (N % 16 == 0)
  const int ntile = NN / 16;
  k_nodes<<<(ntile + NWAVES - 1) / NWAVES, 32 * NWAVES, 0, stream>>>(
      h, h16, agg, indeg, selfh, aggh, selfb, aggb, lng, lnb, (float*)d_out, ntile);
}
